// RecurrentGeneralizedPFAttention_25804163514519
// MI455X (gfx1250) — matrix operand order repaired and checked by reading, not compiled, not run on hardware
//
#include <hip/hip_runtime.h>
#include <stdint.h>

typedef __attribute__((ext_vector_type(16))) _Float16 v16h;
typedef __attribute__((ext_vector_type(8)))  float    v8f;

static constexpr int cB = 8, cN = 1024, cE = 512, cH = 8, cD = 64;
static constexpr int QKV_LD = 3 * cH * cD;   // 1536
static constexpr int HD     = cH * cD;       // 512
static constexpr int KPAD   = 264;           // 256 + 8 pad (f16 LDS panel row)

#define WMMA_F16(a, b, c) \
  __builtin_amdgcn_wmma_f32_16x16x32_f16(false, (a), false, (b), (short)0, (c), false, false)

// ---------------- WMMA fragment builders ----------------
// A-matrix 16x32 f16: lane L holds row L&15; halfs[0..15]=K klo..klo+15,
// klo=(L<16)?0:16.
__device__ __forceinline__ v16h frag_a_f32(const float* __restrict__ src, int ld, int lane) {
  const int m   = lane & 15;
  const int klo = (lane & 16) ? 16 : 0;
  const float* p = src + (size_t)m * ld + klo;
  v16h f;
#pragma unroll
  for (int t = 0; t < 8; ++t) f[t]     = (_Float16)p[t];
#pragma unroll
  for (int t = 0; t < 8; ++t) f[t + 8] = (_Float16)p[t + 8];
  return f;
}

__device__ __forceinline__ v16h frag_a_f16(const _Float16* __restrict__ src, int ld, int lane) {
  const int m   = lane & 15;
  const int klo = (lane & 16) ? 16 : 0;
  const _Float16* p = src + (size_t)m * ld + klo;
  v16h f;
#pragma unroll
  for (int t = 0; t < 8; ++t) f[t]     = p[t];
#pragma unroll
  for (int t = 0; t < 8; ++t) f[t + 8] = p[t + 8];
  return f;
}

// overload dispatch (must precede the template that calls it)
__device__ __forceinline__ v16h frag_a_f16_or_f32(const float* p, int ld, int lane) {
  return frag_a_f32(p, ld, lane);
}
__device__ __forceinline__ v16h frag_a_f16_or_f32(const _Float16* p, int ld, int lane) {
  return frag_a_f16(p, ld, lane);
}

// B-matrix 32x16 where B = tokens^T (f16 tokens row-major [n_tok, K], ld):
// lane L holds column j=L&15 (token row j), 16 consecutive K starting at (L<16)?0:16.
__device__ __forceinline__ v16h frag_b_tok16(const _Float16* __restrict__ tok, int ld, int lane) {
  const int j  = lane & 15;
  const int kb = (lane & 16) ? 16 : 0;
  const _Float16* p = tok + (size_t)j * ld + kb;
  v16h f;
#pragma unroll
  for (int t = 0; t < 16; ++t) f[t] = p[t];
  return f;
}

// B-matrix 32x16 from row-major f16 [K, Nc] (v), leading dim ldb.
__device__ __forceinline__ v16h frag_b_row16(const _Float16* __restrict__ Bm, int ldb, int lane) {
  const int n  = lane & 15;
  const int kb = (lane & 16) ? 16 : 0;
  const _Float16* p = Bm + (size_t)kb * ldb + n;
  v16h f;
#pragma unroll
  for (int t = 0; t < 16; ++t) f[t] = p[(size_t)t * ldb];
  return f;
}

// B fragment from transposed/padded f16 LDS panel [64 cols][KPAD].
__device__ __forceinline__ v16h frag_b_lds16(const _Float16* p, int k0, int jj, int lane) {
  const int c  = (jj << 4) + (lane & 15);
  const int kb = k0 + ((lane & 16) ? 16 : 0);
  const _Float16* q = p + c * KPAD + kb;
  v16h f;
#pragma unroll
  for (int t = 0; t < 16; ++t) f[t] = q[t];
  return f;
}

// ---------------- GEMM  C[M,Nc] = A[M,K] @ B[K,Nc] (+bias on cols < bias_limit) --------
// Block = 8 waves = 128 rows x 64 cols. f32 B panel async-staged to LDS
// (global_load_async_to_lds_b128), converted once to a transposed f16 panel,
// then consumed by all 8 waves (4 accumulators each).
template <typename AT, typename CT>
__global__ __launch_bounds__(256) void gemm_wmma(const AT* __restrict__ A,
                                                 const float* __restrict__ Bm,
                                                 CT* __restrict__ C,
                                                 const float* __restrict__ bias, int bias_limit,
                                                 int M, int Nc, int K) {
  const int lane = threadIdx.x & 31;
  const int wv   = threadIdx.x >> 5;            // 0..7
  const int mblocks = M >> 7;                   // M/128
  const int bm   = blockIdx.x % mblocks;
  const int cg   = blockIdx.x / mblocks;        // 64-col group
  const int col0 = cg * 64;
  const int row0 = bm * 128 + wv * 16;

  __shared__ float    pan32[256 * 64];          // 64 KB raw f32 K-chunk panel
  __shared__ _Float16 pan16[64 * KPAD];         // transposed f16 panel

  v8f acc[4] = {};
  for (int kc = 0; kc < K; kc += 256) {
    __syncthreads();                            // prior-phase consumers done
    {   // async stage B[kc:kc+256, col0:col0+64] -> LDS f32 (16B per lane per op)
      const int c4 = (threadIdx.x & 15) << 2;   // 0,4,...,60
      const int kr = threadIdx.x >> 4;          // 0..15
      const unsigned lbase = (unsigned)(uintptr_t)&pan32[0];
      for (int kk = 0; kk < 256; kk += 16) {
        const int k = kk + kr;
        const unsigned loff = lbase + (unsigned)(((k << 6) + c4) << 2);
        const uint64_t ga = (uint64_t)(uintptr_t)(Bm + (size_t)(kc + k) * Nc + col0 + c4);
        asm volatile("global_load_async_to_lds_b128 %0, %1, off"
                     :: "v"(loff), "v"(ga) : "memory");
      }
      asm volatile("s_wait_asynccnt 0x0" ::: "memory");
    }
    __syncthreads();
    {   // convert once to transposed f16 panel (shared by all 8 waves)
      const int c  = threadIdx.x & 63;
      const int k4 = threadIdx.x >> 6;          // 0..3
      for (int kk = 0; kk < 256; kk += 4) {
        const int k = kk + k4;
        pan16[c * KPAD + k] = (_Float16)pan32[(k << 6) + c];
      }
    }
    __syncthreads();

    const AT* Arow = A + (size_t)row0 * K + kc;
    for (int k0 = 0; k0 < 256; k0 += 32) {
      v16h a = frag_a_f16_or_f32(Arow + k0, K, lane);
#pragma unroll
      for (int jj = 0; jj < 4; ++jj) {
        v16h b = frag_b_lds16(pan16, k0, jj, lane);
        acc[jj] = WMMA_F16(a, b, acc[jj]);
      }
    }
  }

  const int rr = row0 + ((lane >> 4) << 3);
#pragma unroll
  for (int jj = 0; jj < 4; ++jj) {
    const int col = col0 + (jj << 4) + (lane & 15);
    const float bv = (bias && col < bias_limit) ? bias[col] : 0.0f;
#pragma unroll
    for (int r = 0; r < 8; ++r)
      C[(size_t)(rr + r) * Nc + col] = (CT)(acc[jj][r] + bv);
  }
}

// ---------------- AC scores -> dots (straight into d_out attn region) ----------
// qkv16 holds q (pre-biased with bias_time_e), k, v as f16. One wave = 16x64 strip.
__global__ __launch_bounds__(256) void attn_scores_ac(const _Float16* __restrict__ qkv16,
                                                      float* __restrict__ dots) {
  const int wid  = (int)((blockIdx.x * blockDim.x + threadIdx.x) >> 5);
  const int lane = threadIdx.x & 31;
  const int jg = wid & 15;
  const int it = (wid >> 4) & 63;
  const int bh = wid >> 10;
  if (bh >= cB * cH) return;
  const int b = bh / cH, h = bh % cH;

  const _Float16* qbase = qkv16 + (size_t)(b * cN + it * 16) * QKV_LD + h * cD;
  const _Float16* kbase = qkv16 + (size_t)(b * cN + jg * 64) * QKV_LD + HD + h * cD;

  v8f acc[4] = {};
#pragma unroll
  for (int k0 = 0; k0 < cD; k0 += 32) {
    v16h a = frag_a_f16(qbase + k0, QKV_LD, lane);
#pragma unroll
    for (int jj = 0; jj < 4; ++jj) {
      v16h bf = frag_b_tok16(kbase + (size_t)(jj * 16) * QKV_LD + k0, QKV_LD, lane);
      acc[jj] = WMMA_F16(a, bf, acc[jj]);
    }
  }
  const size_t base = (size_t)bh * cN * cN;
  const int i0 = it * 16 + ((lane >> 4) << 3);
#pragma unroll
  for (int jj = 0; jj < 4; ++jj) {
    const int j = jg * 64 + jj * 16 + (lane & 15);
#pragma unroll
    for (int r = 0; r < 8; ++r)
      dots[base + (size_t)(i0 + r) * cN + j] = acc[jj][r];
  }
}

// ---------------- BD_t + BD_c with fused rel_shift scatter-add ----------------
// BD[a,c] = qb[a].wr[c]; injective inverse of _rel_shift:
//   c >= N-1-a          -> dots[a,   c-(N-1)+a] += 0.5*val
//   c <= N-2-a && a>=1  -> dots[a-1, c+a+1    ] += 0.5*val
// slot (i,i+1) receives nothing (the inserted zero). 8 WMMAs per A-load.
__global__ __launch_bounds__(256) void attn_scores_bd(const _Float16* __restrict__ qkv16,
                                                      const _Float16* __restrict__ wrt16,
                                                      const _Float16* __restrict__ wrc16,
                                                      float* __restrict__ dots) {
  const int wid  = (int)((blockIdx.x * blockDim.x + threadIdx.x) >> 5);
  const int lane = threadIdx.x & 31;
  const int jg = wid & 15;
  const int it = (wid >> 4) & 63;
  const int bh = wid >> 10;
  if (bh >= cB * cH) return;
  const int b = bh / cH, h = bh % cH;

  const _Float16* qbase = qkv16 + (size_t)(b * cN + it * 16) * QKV_LD + h * cD;
  const _Float16* tbase = wrt16 + (size_t)(b * cN + jg * 64) * HD + h * cD;
  const _Float16* cbase = wrc16 + (size_t)(b * cN + jg * 64) * HD + h * cD;

  v8f accT[4] = {}, accC[4] = {};
#pragma unroll
  for (int k0 = 0; k0 < cD; k0 += 32) {
    v16h a = frag_a_f16(qbase + k0, QKV_LD, lane);
#pragma unroll
    for (int jj = 0; jj < 4; ++jj) {
      v16h b1 = frag_b_tok16(tbase + (size_t)(jj * 16) * HD + k0, HD, lane);
      v16h b2 = frag_b_tok16(cbase + (size_t)(jj * 16) * HD + k0, HD, lane);
      accT[jj] = WMMA_F16(a, b1, accT[jj]);
      accC[jj] = WMMA_F16(a, b2, accC[jj]);
    }
  }
  const size_t base = (size_t)bh * cN * cN;
  const int a0 = it * 16 + ((lane >> 4) << 3);
#pragma unroll
  for (int jj = 0; jj < 4; ++jj) {
    const int cc = jg * 64 + jj * 16 + (lane & 15);
#pragma unroll
    for (int r = 0; r < 8; ++r) {
      const int arow = a0 + r;
      const float val = 0.5f * (accT[jj][r] + accC[jj][r]);
      if (cc >= cN - 1 - arow) {
        const int j = cc - (cN - 1) + arow;
        dots[base + (size_t)arow * cN + j] += val;
      } else if (arow >= 1) {
        const int j = cc + arow + 1;
        dots[base + (size_t)(arow - 1) * cN + j] += val;
      }
    }
  }
}

// ---------------- softmax over HEADS (torch dim=-1 on [i,j,b,h]) in place -------
__global__ __launch_bounds__(256) void softmax_heads(float* __restrict__ attn) {
  const size_t t  = (size_t)blockIdx.x * blockDim.x + threadIdx.x;
  const size_t p4 = t * 4;
  if (p4 >= (size_t)cB * cN * cN) return;
  const size_t b  = p4 / ((size_t)cN * cN);
  const size_t ij = p4 % ((size_t)cN * cN);
  float* base = attn + b * (size_t)cH * cN * cN + ij;

  float v[cH][4];
#pragma unroll
  for (int h = 0; h < cH; ++h) {
    const float4 x4 = *(const float4*)(base + (size_t)h * cN * cN);
    v[h][0] = x4.x; v[h][1] = x4.y; v[h][2] = x4.z; v[h][3] = x4.w;
  }
#pragma unroll
  for (int c = 0; c < 4; ++c) {
    float m = v[0][c];
#pragma unroll
    for (int h = 1; h < cH; ++h) m = fmaxf(m, v[h][c]);
    float s = 0.0f;
#pragma unroll
    for (int h = 0; h < cH; ++h) { const float e = __expf(v[h][c] - m); v[h][c] = e; s += e; }
    const float inv = 1.0f / s;
#pragma unroll
    for (int h = 0; h < cH; ++h) v[h][c] *= inv;
  }
#pragma unroll
  for (int h = 0; h < cH; ++h) {
    float4 x4; x4.x = v[h][0]; x4.y = v[h][1]; x4.z = v[h][2]; x4.w = v[h][3];
    *(float4*)(base + (size_t)h * cN * cN) = x4;
  }
}

// ---------------- ctx16[(b,i),(h,d)] = sum_j attn[b,h,i,j] * v[j,b,h,d] -----------
__global__ __launch_bounds__(256) void attn_ctx(const float* __restrict__ attn,
                                                const _Float16* __restrict__ qkv16,
                                                _Float16* __restrict__ ctx16) {
  const int wid  = (int)((blockIdx.x * blockDim.x + threadIdx.x) >> 5);
  const int lane = threadIdx.x & 31;
  const int it = wid & 63;
  const int bh = wid >> 6;
  if (bh >= cB * cH) return;
  const int b = bh / cH, h = bh % cH;

  const float*    abase = attn + (size_t)bh * cN * cN + (size_t)(it * 16) * cN;  // ld = N
  const _Float16* vbase = qkv16 + (size_t)b * cN * QKV_LD + 2 * HD + h * cD;     // v cols

  v8f acc[4] = {};
  for (int j0 = 0; j0 < cN; j0 += 32) {
    v16h a = frag_a_f32(abase + j0, cN, lane);
#pragma unroll
    for (int dt = 0; dt < 4; ++dt) {
      v16h bf = frag_b_row16(vbase + (size_t)j0 * QKV_LD + dt * 16, QKV_LD, lane);
      acc[dt] = WMMA_F16(a, bf, acc[dt]);
    }
  }
  const int i0 = it * 16 + ((lane >> 4) << 3);
#pragma unroll
  for (int dt = 0; dt < 4; ++dt) {
    const int col = h * cD + dt * 16 + (lane & 15);
#pragma unroll
    for (int r = 0; r < 8; ++r)
      ctx16[(size_t)(b * cN + i0 + r) * HD + col] = (_Float16)acc[dt][r];
  }
}

// ---------------- launch ----------------
extern "C" void kernel_launch(void* const* d_in, const int* in_sizes, int n_in,
                              void* d_out, int out_size, void* d_ws, size_t ws_size,
                              hipStream_t stream) {
  const float* x       = (const float*)d_in[0];
  const float* r_t     = (const float*)d_in[1];
  const float* r_c     = (const float*)d_in[2];
  const float* bias_te = (const float*)d_in[3];
  // d_in[4..6] are unused by the reference math
  const float* w_qkv   = (const float*)d_in[7];
  const float* w_kr_t  = (const float*)d_in[8];
  const float* w_kr_c  = (const float*)d_in[9];
  const float* w_out   = (const float*)d_in[10];
  const float* b_out   = (const float*)d_in[11];

  float* out  = (float*)d_out;                              // [B,N,E]
  float* attn = (float*)d_out + (size_t)cB * cN * cE;       // [B,H,N,N] dots->softmax in place

  _Float16* h16   = (_Float16*)d_ws;
  _Float16* qkv16 = h16;                                    // [8192,1536] f16 (q pre-biased)
  _Float16* wrt16 = qkv16 + (size_t)cB * cN * QKV_LD;       // [8192,512] f16
  _Float16* wrc16 = wrt16 + (size_t)cB * cN * HD;           // [8192,512] f16
  _Float16* ctx16 = wrc16 + (size_t)cB * cN * HD;           // [8192,512] f16

  const int M = cB * cN;                                    // 8192
  dim3 blk(256);

  // 1) projections -> f16 activations (qkv folds bias_time_e into q columns)
  gemm_wmma<float, _Float16><<<dim3((M / 128) * (QKV_LD / 64)), blk, 0, stream>>>(
      x, w_qkv, qkv16, bias_te, HD, M, QKV_LD, cE);
  gemm_wmma<float, _Float16><<<dim3((M / 128) * (HD / 64)), blk, 0, stream>>>(
      r_t, w_kr_t, wrt16, nullptr, 0, M, HD, cE);
  gemm_wmma<float, _Float16><<<dim3((M / 128) * (HD / 64)), blk, 0, stream>>>(
      r_c, w_kr_c, wrc16, nullptr, 0, M, HD, cE);

  // 2) AC scores into dots (= attn slot of d_out)
  const int score_waves = cB * cH * (cN / 16) * (cN / 64);  // 65536
  attn_scores_ac<<<dim3(score_waves / 8), blk, 0, stream>>>(qkv16, attn);

  // 3) BD_t+BD_c with fused rel_shift scatter-add
  attn_scores_bd<<<dim3(score_waves / 8), blk, 0, stream>>>(qkv16, wrt16, wrc16, attn);

  // 4) softmax over heads, in place
  const size_t sm_threads = (size_t)cB * cN * cN / 4;       // 2M
  softmax_heads<<<dim3((unsigned)(sm_threads / 256)), blk, 0, stream>>>(attn);

  // 5) attn @ v -> f16 ctx
  const int ctx_waves = cB * cH * (cN / 16);                // 4096
  attn_ctx<<<dim3(ctx_waves / 8), blk, 0, stream>>>(attn, qkv16, ctx16);

  // 6) output projection + bias (f16 A, f32 C)
  gemm_wmma<_Float16, float><<<dim3((M / 128) * (HD / 64)), blk, 0, stream>>>(
      ctx16, w_out, out, b_out, HD, M, cE, HD);
}
